// MHA_SP2NormFused_88098369175554
// MI455X (gfx1250) — matrix operand order repaired and checked by reading, not compiled, not run on hardware
//
#include <hip/hip_runtime.h>
#include <hip/hip_bf16.h>

typedef __attribute__((ext_vector_type(16))) _Float16 v16h;
typedef __attribute__((ext_vector_type(8)))  _Float16 v8h;
typedef __attribute__((ext_vector_type(8)))  float    v8f;

#define H_    12
#define T_    2048
#define C_    768
#define HS_   64
#define BT_   4096     // B*T
#define N3C_  2304     // 3*C

// ---------------- CDNA5 async copy helpers ----------------

// global_load_async_to_lds_b128: per-lane 16B DMA global -> LDS (ASYNCcnt).
// GVS addressing: mem = SGPR_base + VGPR_offset. VDST carries the LDS byte addr.
__device__ __forceinline__ void async_ld16(void* lds, const void* gbase, unsigned voff_bytes) {
  unsigned lds_off = (unsigned)(size_t)lds;   // low 32 bits of generic ptr = LDS offset
  asm volatile("global_load_async_to_lds_b128 %0, %1, %2"
               :: "v"(lds_off), "v"(voff_bytes),
                  "s"((unsigned long long)(size_t)gbase)
               : "memory");
}
__device__ __forceinline__ void wait_async0() {
  asm volatile("s_wait_asynccnt 0" ::: "memory");
}

// ---------------- WMMA helpers (gfx1250 wave32) ----------------

__device__ __forceinline__ v8f wmma16(v16h a, v16h b, v8f c) {
  return __builtin_amdgcn_wmma_f32_16x16x32_f16(false, a, false, b, (short)0, c,
                                                false, false);
}

// A fragment (16-bit A 16x32, MxK) from row-major LDS tile [M][stride].
__device__ __forceinline__ v16h load_a_frag(const _Float16* src, int stride) {
  int lane = threadIdx.x & 31;
  const _Float16* p = src + (lane & 15) * stride + ((lane >> 4) << 3);
  v16h a;
#pragma unroll
  for (int e = 0; e < 8; ++e) a[e] = p[e];
#pragma unroll
  for (int e = 0; e < 8; ++e) a[8 + e] = p[16 + e];
  return a;
}

// B fragment (32x16 KxN) from LDS stored as [N][K] (contiguous along K).
__device__ __forceinline__ v16h load_b_frag_nk(const _Float16* src, int stride) {
  int lane = threadIdx.x & 31;
  const _Float16* p = src + (lane & 15) * stride + ((lane >> 4) << 3);
  v16h b;
#pragma unroll
  for (int e = 0; e < 8; ++e) b[e] = p[e];
#pragma unroll
  for (int e = 0; e < 8; ++e) b[8 + e] = p[16 + e];
  return b;
}

// ---------------- fp32 -> f16 cast (row-major copy) ----------------

__global__ void __launch_bounds__(256)
cast_f32_f16(const float* __restrict__ s, _Float16* __restrict__ d, int n) {
  int i = (blockIdx.x * 256 + threadIdx.x) * 8;
  if (i + 8 <= n) {
    float4 a = *(const float4*)(s + i);
    float4 b = *(const float4*)(s + i + 4);
    v8h o;
    o[0] = (_Float16)a.x; o[1] = (_Float16)a.y;
    o[2] = (_Float16)a.z; o[3] = (_Float16)a.w;
    o[4] = (_Float16)b.x; o[5] = (_Float16)b.y;
    o[6] = (_Float16)b.z; o[7] = (_Float16)b.w;
    *(v8h*)(d + i) = o;
  }
}

// ---------------- fp32 [K][N] -> f16 [N][K] transpose (for weights) ----------------

__global__ void __launch_bounds__(256)
cast_transpose(const float* __restrict__ W, _Float16* __restrict__ WT, int K, int N) {
  __shared__ _Float16 t[32][33];
  const int k0 = blockIdx.y * 32, n0 = blockIdx.x * 32;
  const int tx = threadIdx.x & 31, ty = threadIdx.x >> 5;
#pragma unroll
  for (int r = ty; r < 32; r += 8)
    t[r][tx] = (_Float16)W[(size_t)(k0 + r) * N + n0 + tx];
  __syncthreads();
#pragma unroll
  for (int r = ty; r < 32; r += 8)
    WT[(size_t)(n0 + r) * K + k0 + tx] = t[tx][r];
}

// ---------------- QKV GEMM (async double-buffered) ----------------
// X[4096x768] @ Wqkv -> q,k [bh][t][d] f16 ; v stored transposed [bh][d][t] f16.
// WT is Wqkv transposed: [2304][768].

__global__ void __launch_bounds__(128)
qkv_gemm(const _Float16* __restrict__ X, const _Float16* __restrict__ WT,
         _Float16* __restrict__ Q, _Float16* __restrict__ K, _Float16* __restrict__ VT) {
  __shared__ _Float16 As[2][64][40];   // [m][k]
  __shared__ _Float16 Bs[2][64][40];   // [n][k]
  __shared__ _Float16 Ts[64][72];      // V-tile transpose staging

  const int m0 = blockIdx.y * 64;
  const int n0 = blockIdx.x * 64;
  const int tid = threadIdx.x, wave = tid >> 5, lane = tid & 31;

  auto stage = [&](int buf, int k0) {
#pragma unroll
    for (int j = 0; j < 2; ++j) {
      int c = tid * 2 + j;               // 0..255
      int row = c >> 2, col = (c & 3) * 8;
      async_ld16(&As[buf][row][col], X,  (unsigned)(((m0 + row) * C_ + k0 + col) * 2));
      async_ld16(&Bs[buf][row][col], WT, (unsigned)(((n0 + row) * C_ + k0 + col) * 2));
    }
  };

  stage(0, 0);
  v8f acc[4] = {};

  for (int i = 0; i < C_ / 32; ++i) {
    wait_async0();
    __syncthreads();
    if (i + 1 < C_ / 32) stage((i + 1) & 1, (i + 1) * 32);
    const int buf = i & 1;
    v16h a = load_a_frag(&As[buf][wave * 16][0], 40);
#pragma unroll
    for (int nt = 0; nt < 4; ++nt)
      acc[nt] = wmma16(a, load_b_frag_nk(&Bs[buf][nt * 16][0], 40), acc[nt]);
  }

  const int three = n0 / C_;
  const int h     = (n0 % C_) / HS_;
  const int b     = m0 / T_;
  const int t0    = m0 - b * T_;
  const int nlo   = lane & 15;
  const int mh    = (lane >> 4) << 3;

  if (three == 2) {
    // V: transpose tile via LDS, store [bh][d][t] coalesced
#pragma unroll
    for (int nt = 0; nt < 4; ++nt)
#pragma unroll
      for (int r = 0; r < 8; ++r)
        Ts[nt * 16 + nlo][wave * 16 + mh + r] = (_Float16)acc[nt][r];
    __syncthreads();
    const int drow = tid >> 1, c0 = (tid & 1) * 32;
    _Float16* vp = VT + (size_t)(b * H_ + h) * HS_ * T_ + (size_t)drow * T_ + t0 + c0;
#pragma unroll
    for (int j = 0; j < 4; ++j)
      *(v8h*)(vp + j * 8) = *(const v8h*)&Ts[drow][c0 + j * 8];
  } else {
    _Float16* outp = (three == 0 ? Q : K) + (size_t)(b * H_ + h) * T_ * HS_;
    const int tbase = t0 + wave * 16 + mh;
#pragma unroll
    for (int nt = 0; nt < 4; ++nt) {
      int d = nt * 16 + nlo;
#pragma unroll
      for (int r = 0; r < 8; ++r)
        outp[(size_t)(tbase + r) * HS_ + d] = (_Float16)acc[nt][r];
    }
  }
}

// ---------------- Attention (streaming softplus-norm, async double-buffered K/V) ----------------
// Q,K: [bh][t][d] f16 ; VT: [bh][d][t] f16 ; O: [B][T][C] f16.

__global__ void __launch_bounds__(128)
attn_sp2norm(const _Float16* __restrict__ Q, const _Float16* __restrict__ K,
             const _Float16* __restrict__ VT, _Float16* __restrict__ O) {
  __shared__ _Float16 Qs[64][72];         // [q][d]
  __shared__ _Float16 Ks[2][64][72];      // [key][d]   == [N][K] for scores
  __shared__ _Float16 Vs[2][64][72];      // [d][key]   == [N][K] for w@V
  __shared__ _Float16 Ws[4][16][72];      // per-wave w tile [m][key]

  const int qt = blockIdx.x;
  const int bh = blockIdx.y;
  const _Float16* Kb = K  + (size_t)bh * T_ * HS_;
  const _Float16* Vb = VT + (size_t)bh * HS_ * T_;
  const _Float16* Qb = Q  + (size_t)bh * T_ * HS_;
  const int tid = threadIdx.x, wave = tid >> 5, lane = tid & 31;
  const int nlo = lane & 15;
  const int mh  = (lane >> 4) << 3;

  auto stage_kv = [&](int buf, int kt) {
#pragma unroll
    for (int j = 0; j < 4; ++j) {
      int c = tid * 4 + j;                 // 0..511
      int row = c >> 3, col = (c & 7) * 8;
      async_ld16(&Ks[buf][row][col], Kb, (unsigned)((((kt * 64 + row) * HS_) + col) * 2));
      async_ld16(&Vs[buf][row][col], Vb, (unsigned)(((row * T_) + kt * 64 + col) * 2));
    }
  };

  { // stage Q tile + first K/V tile
#pragma unroll
    for (int j = 0; j < 4; ++j) {
      int c = tid * 4 + j;
      int row = c >> 3, col = (c & 7) * 8;
      async_ld16(&Qs[row][col], Qb, (unsigned)((((qt * 64 + row) * HS_) + col) * 2));
    }
    stage_kv(0, 0);
  }
  wait_async0();
  __syncthreads();

  v16h qa0 = load_a_frag(&Qs[wave * 16][0], 72);
  v16h qa1 = load_a_frag(&Qs[wave * 16][32], 72);

  v8f  o[4] = {};
  float den[8] = {};
  const int qrow0 = qt * 64 + wave * 16 + mh;

  for (int kt = 0; kt <= qt; ++kt) {
    if (kt > 0) { wait_async0(); __syncthreads(); }
    if (kt < qt) stage_kv((kt + 1) & 1, kt + 1);
    const int buf = kt & 1;
    const bool diag = (kt == qt);

#pragma unroll
    for (int nt = 0; nt < 4; ++nt) {
      v8f s = {};
      s = wmma16(qa0, load_b_frag_nk(&Ks[buf][nt * 16][0], 72), s);
      s = wmma16(qa1, load_b_frag_nk(&Ks[buf][nt * 16][32], 72), s);
      const int key = kt * 64 + nt * 16 + nlo;
#pragma unroll
      for (int r = 0; r < 8; ++r) {
        float sc = s[r] * 0.125f;                         // HS^-0.5
        float w  = (sc > 15.f) ? sc : log1pf(__expf(sc)); // softplus
        if (diag && key > qrow0 + r) w = 0.f;
        den[r] += w;
        Ws[wave][mh + r][nt * 16 + nlo] = (_Float16)w;
      }
    }
    __syncthreads();   // cross-lane Ws visibility

#pragma unroll
    for (int ks = 0; ks < 2; ++ks) {
      v16h wa = load_a_frag(&Ws[wave][0][ks * 32], 72);
#pragma unroll
      for (int nt = 0; nt < 4; ++nt)
        o[nt] = wmma16(wa, load_b_frag_nk(&Vs[buf][nt * 16][ks * 32], 72), o[nt]);
    }
  }

#pragma unroll
  for (int r = 0; r < 8; ++r) {
    float d = den[r];
    d += __shfl_xor(d, 1, 32);
    d += __shfl_xor(d, 2, 32);
    d += __shfl_xor(d, 4, 32);
    d += __shfl_xor(d, 8, 32);
    den[r] = d;
  }

  const int b = bh / H_, h = bh % H_;
#pragma unroll
  for (int nt = 0; nt < 4; ++nt) {
    int d = nt * 16 + nlo;
#pragma unroll
    for (int r = 0; r < 8; ++r) {
      int t = qt * 64 + wave * 16 + mh + r;
      O[(size_t)(b * T_ + t) * C_ + h * HS_ + d] = (_Float16)(o[nt][r] / den[r]);
    }
  }
}

// ---------------- Proj GEMM (async double-buffered): A @ Wproj^T(stored [N][K]) + bias ----------------

__global__ void __launch_bounds__(128)
proj_gemm(const _Float16* __restrict__ A, const _Float16* __restrict__ WT,
          const float* __restrict__ bias, float* __restrict__ out) {
  __shared__ _Float16 As[2][64][40];
  __shared__ _Float16 Bs[2][64][40];

  const int m0 = blockIdx.y * 64;
  const int n0 = blockIdx.x * 64;
  const int tid = threadIdx.x, wave = tid >> 5, lane = tid & 31;

  auto stage = [&](int buf, int k0) {
#pragma unroll
    for (int j = 0; j < 2; ++j) {
      int c = tid * 2 + j;
      int row = c >> 2, col = (c & 3) * 8;
      async_ld16(&As[buf][row][col], A,  (unsigned)(((m0 + row) * C_ + k0 + col) * 2));
      async_ld16(&Bs[buf][row][col], WT, (unsigned)(((n0 + row) * C_ + k0 + col) * 2));
    }
  };

  stage(0, 0);
  v8f acc[4] = {};

  for (int i = 0; i < C_ / 32; ++i) {
    wait_async0();
    __syncthreads();
    if (i + 1 < C_ / 32) stage((i + 1) & 1, (i + 1) * 32);
    const int buf = i & 1;
    v16h a = load_a_frag(&As[buf][wave * 16][0], 40);
#pragma unroll
    for (int nt = 0; nt < 4; ++nt)
      acc[nt] = wmma16(a, load_b_frag_nk(&Bs[buf][nt * 16][0], 40), acc[nt]);
  }

  const int mbase = m0 + wave * 16 + ((lane >> 4) << 3);
  const int nlo = lane & 15;
#pragma unroll
  for (int nt = 0; nt < 4; ++nt) {
    int n = n0 + nt * 16 + nlo;
    float bv = bias[n];
#pragma unroll
    for (int r = 0; r < 8; ++r)
      out[(size_t)(mbase + r) * C_ + n] = acc[nt][r] + bv;
  }
}

// ---------------- host launch ----------------

extern "C" void kernel_launch(void* const* d_in, const int* in_sizes, int n_in,
                              void* d_out, int out_size, void* d_ws, size_t ws_size,
                              hipStream_t stream) {
  const float* x     = (const float*)d_in[0];   // [B,T,C]
  const float* Wqkv  = (const float*)d_in[1];   // [C,3C]
  const float* Wproj = (const float*)d_in[2];   // [C,C]
  const float* bproj = (const float*)d_in[3];   // [C]
  float* out = (float*)d_out;

  char* ws = (char*)d_ws;
  size_t off = 0;
  auto carve = [&](size_t nhalf) -> _Float16* {
    _Float16* p = (_Float16*)(ws + off);
    off += ((nhalf * sizeof(_Float16)) + 255) & ~(size_t)255;
    return p;
  };
  _Float16* x16  = carve((size_t)BT_ * C_);
  _Float16* wqT  = carve((size_t)N3C_ * C_);         // Wqkv^T  [3C][C]
  _Float16* wpT  = carve((size_t)C_ * C_);           // Wproj^T [C][C]
  _Float16* q16  = carve((size_t)2 * H_ * T_ * HS_); // [bh][t][d]
  _Float16* k16  = carve((size_t)2 * H_ * T_ * HS_); // [bh][t][d]
  _Float16* vT16 = carve((size_t)2 * H_ * HS_ * T_); // [bh][d][t]
  _Float16* a16  = carve((size_t)BT_ * C_);          // [B*T][C]

  const int nx = BT_ * C_;
  cast_f32_f16<<<nx / 2048, 256, 0, stream>>>(x, x16, nx);
  cast_transpose<<<dim3(N3C_ / 32, C_ / 32), 256, 0, stream>>>(Wqkv, wqT, C_, N3C_);
  cast_transpose<<<dim3(C_ / 32, C_ / 32), 256, 0, stream>>>(Wproj, wpT, C_, C_);

  qkv_gemm<<<dim3(N3C_ / 64, BT_ / 64), 128, 0, stream>>>(x16, wqT, q16, k16, vT16);
  attn_sp2norm<<<dim3(T_ / 64, 2 * H_), 128, 0, stream>>>(q16, k16, vT16, a16);
  proj_gemm<<<dim3(C_ / 64, BT_ / 64), 128, 0, stream>>>(a16, wpT, bproj, out);
}
